// MultiplicativeAttention_74268574482974
// MI455X (gfx1250) — matrix operand order repaired and checked by reading, not compiled, not run on hardware
//
#include <hip/hip_runtime.h>

// ---------------------------------------------------------------------------
// MultiplicativeAttention for MI455X (gfx1250, wave32, WMMA bf16 path)
//   attn_w  = softmax(mask(q @ w @ s^T))          [B,2048,2048] f32 (output 0)
//   attn_out= (attn_w @ s) @ out_w^T + out_b      [B,2048, 512] f32 (output 1)
// All big GEMMs run on v_wmma_f32_16x16x32_bf16 with f32 accumulation.
// ---------------------------------------------------------------------------

typedef __attribute__((ext_vector_type(16))) __bf16 v16bf;
typedef __attribute__((ext_vector_type(8)))  __bf16 v8bf;
typedef __attribute__((ext_vector_type(8)))  float  v8f;
typedef __attribute__((ext_vector_type(4)))  float  v4f;

#define LDST 40               // padded LDS row stride (bf16 elems) = 80B
#define NEG_INF_F (-10000000000.0f)

// ---- stage a 128x32 tile (rows x K) into LDS, converting f32->bf16 if needed
template <bool F32>
__device__ __forceinline__ void stage_tile(const char* __restrict__ base,
                                           int row0, int k0, int ld,
                                           __bf16* __restrict__ dst, int tid) {
  const int r = tid >> 1;               // 128 rows, 2 threads per row
  const int c = (tid & 1) << 4;         // 16 elements per thread
  const size_t off = (size_t)(row0 + r) * (size_t)ld + (size_t)(k0 + c);
  __bf16* d = dst + r * LDST + c;
  if (F32) {
    const float* p = (const float*)base + off;
#pragma unroll
    for (int i = 0; i < 16; i += 4) {
      v4f f = *(const v4f*)(p + i);
      d[i + 0] = (__bf16)f[0];
      d[i + 1] = (__bf16)f[1];
      d[i + 2] = (__bf16)f[2];
      d[i + 3] = (__bf16)f[3];
    }
  } else {
    const v8bf* p = (const v8bf*)((const __bf16*)base + off);
    *(v8bf*)(d)     = p[0];
    *(v8bf*)(d + 8) = p[1];
  }
}

// ---- WMMA A fragment (16x32 bf16) from LDS tile
// lane 0-15: M=lane, K in {0..7, 16..23}; lane 16-31: M=lane-16, K in {8..15, 24..31}
__device__ __forceinline__ v16bf load_frag_a(const __bf16* __restrict__ tile,
                                             int row0, int lane) {
  const __bf16* p = tile + (row0 + (lane & 15)) * LDST + ((lane >> 4) << 3);
  v8bf lo = *(const v8bf*)p;            // K = kb .. kb+7
  v8bf hi = *(const v8bf*)(p + 16);     // K = kb+16 .. kb+23
  v16bf r;
#pragma unroll
  for (int i = 0; i < 8; ++i) { r[i] = lo[i]; r[i + 8] = hi[i]; }
  return r;
}

// ---- WMMA B fragment (32x16 bf16) from LDS tile holding B^T rows (N x K)
// lane 0-15: N=lane, K in {0..7, 16..23}; lane 16-31: N=lane-16, K in {8..15, 24..31}
__device__ __forceinline__ v16bf load_frag_b(const __bf16* __restrict__ tile,
                                             int col0, int lane) {
  const __bf16* p = tile + (col0 + (lane & 15)) * LDST + ((lane >> 4) << 3);
  v8bf lo = *(const v8bf*)p;
  v8bf hi = *(const v8bf*)(p + 16);
  v16bf r;
#pragma unroll
  for (int i = 0; i < 8; ++i) { r[i] = lo[i]; r[i + 8] = hi[i]; }
  return r;
}

// ---------------------------------------------------------------------------
// Block-tiled GEMM: C[M,N] = A[M,K] * Bt[N,K]^T  (Bt stored row-major N x K)
// Block = 128x128, 256 threads = 8 waves (4 over M x 2 over N),
// wave tile 32x64 = 2x4 WMMA 16x16 accumulators. Double-buffered LDS.
// EPI: 1 = mask->NEG_INF, store f32 | 2 = store bf16 | 3 = +bias, store f32
// ---------------------------------------------------------------------------
template <bool AF32, bool BF32, int EPI>
__global__ __launch_bounds__(256) void gemm_bf16_wmma(
    const char* __restrict__ A, const char* __restrict__ Bt,
    char* __restrict__ C, int N, int K, long batchA, long batchB, long batchC,
    const int* __restrict__ mask, int maskStride,
    const float* __restrict__ bias) {
  __shared__ __bf16 As[2][128 * LDST];
  __shared__ __bf16 Bs[2][128 * LDST];

  const int tid  = threadIdx.x;
  const int lane = tid & 31;
  const int wave = tid >> 5;
  const int wm = wave & 3;              // 32-row slice within 128
  const int wn = wave >> 2;             // 64-col slice within 128
  const int z = blockIdx.z;
  const int rowBase = blockIdx.y << 7;
  const int colBase = blockIdx.x << 7;

  const char* Ab = A + (size_t)z * (size_t)batchA * (AF32 ? 4u : 2u);
  const char* Bb = Bt + (size_t)z * (size_t)batchB * (BF32 ? 4u : 2u);
  char* Cb = C + (size_t)z * (size_t)batchC * (EPI == 2 ? 2u : 4u);

  v8f acc[2][4];
#pragma unroll
  for (int mt = 0; mt < 2; ++mt)
#pragma unroll
    for (int nt = 0; nt < 4; ++nt)
#pragma unroll
      for (int r = 0; r < 8; ++r) acc[mt][nt][r] = 0.0f;

  const int nk = K >> 5;
  stage_tile<AF32>(Ab, rowBase, 0, K, As[0], tid);
  stage_tile<BF32>(Bb, colBase, 0, K, Bs[0], tid);
  __syncthreads();

  for (int kt = 0; kt < nk; ++kt) {
    const int buf = kt & 1;
    if (kt + 1 < nk) {                  // prefetch next K-slice into other buffer
      stage_tile<AF32>(Ab, rowBase, (kt + 1) << 5, K, As[buf ^ 1], tid);
      stage_tile<BF32>(Bb, colBase, (kt + 1) << 5, K, Bs[buf ^ 1], tid);
    }
    v16bf afr[2], bfr[4];
#pragma unroll
    for (int mt = 0; mt < 2; ++mt)
      afr[mt] = load_frag_a(As[buf], (wm << 5) + (mt << 4), lane);
#pragma unroll
    for (int nt = 0; nt < 4; ++nt)
      bfr[nt] = load_frag_b(Bs[buf], (wn << 6) + (nt << 4), lane);
#pragma unroll
    for (int mt = 0; mt < 2; ++mt)
#pragma unroll
      for (int nt = 0; nt < 4; ++nt)
        acc[mt][nt] = __builtin_amdgcn_wmma_f32_16x16x32_bf16(
            false, afr[mt], false, bfr[nt], (short)0, acc[mt][nt], false, false);
    __syncthreads();
  }

  // Epilogue: C/D layout -> lane n = lane&15, VGPR r holds M = r + 8*(lane>>4)
  const int lrow = (wm << 5) + ((lane >> 4) << 3);
  const int lcol = (wn << 6) + (lane & 15);
#pragma unroll
  for (int nt = 0; nt < 4; ++nt) {
    const int gcol = colBase + lcol + (nt << 4);
    int mk = 1;
    float bs = 0.0f;
    if (EPI == 1) mk = mask[(size_t)z * maskStride + gcol];
    if (EPI == 3) bs = bias[gcol];
#pragma unroll
    for (int mt = 0; mt < 2; ++mt) {
#pragma unroll
      for (int r = 0; r < 8; ++r) {
        const int grow = rowBase + lrow + (mt << 4) + r;
        float v = acc[mt][nt][r];
        if (EPI == 1) v = mk ? v : NEG_INF_F;
        if (EPI == 3) v += bs;
        const size_t idx = (size_t)grow * (size_t)N + (size_t)gcol;
        if (EPI == 2) ((__bf16*)Cb)[idx] = (__bf16)v;
        else          ((float*)Cb)[idx] = v;
      }
    }
  }
}

// ---------------------------------------------------------------------------
// Transpose + convert: out[c*rows + r] = (bf16) in[r*cols + c]
// ---------------------------------------------------------------------------
__global__ __launch_bounds__(256) void transpose_f32_bf16(
    const float* __restrict__ in, __bf16* __restrict__ out, int rows, int cols,
    long sIn, long sOut) {
  __shared__ float tile[32][33];
  const int z = blockIdx.z;
  const float* ip = in + (size_t)z * (size_t)sIn;
  __bf16* op = out + (size_t)z * (size_t)sOut;
  const int r0 = blockIdx.y << 5, c0 = blockIdx.x << 5;
  const int tx = threadIdx.x & 31, ty = threadIdx.x >> 5;
#pragma unroll
  for (int i = 0; i < 32; i += 8)
    tile[ty + i][tx] = ip[(size_t)(r0 + ty + i) * cols + (c0 + tx)];
  __syncthreads();
#pragma unroll
  for (int i = 0; i < 32; i += 8)
    op[(size_t)(c0 + ty + i) * rows + (r0 + tx)] = (__bf16)tile[tx][ty + i];
}

// ---------------------------------------------------------------------------
// In-place row softmax over 2048 elements; one 256-thread block per row.
// ---------------------------------------------------------------------------
__global__ __launch_bounds__(256) void softmax_row2048(float* __restrict__ data) {
  float* row = data + (size_t)blockIdx.x * 2048u;
  const int tid = threadIdx.x;
  const int lane = tid & 31, wave = tid >> 5;
  __shared__ float smax[8];
  __shared__ float ssum[8];

  float v[8];
  float m = -3.4e38f;
#pragma unroll
  for (int i = 0; i < 8; ++i) {
    v[i] = row[tid + (i << 8)];
    m = fmaxf(m, v[i]);
  }
#pragma unroll
  for (int off = 16; off; off >>= 1) m = fmaxf(m, __shfl_xor(m, off, 32));
  if (lane == 0) smax[wave] = m;
  __syncthreads();
  m = smax[0];
#pragma unroll
  for (int i = 1; i < 8; ++i) m = fmaxf(m, smax[i]);

  float s = 0.0f;
#pragma unroll
  for (int i = 0; i < 8; ++i) {
    v[i] = __expf(v[i] - m);
    s += v[i];
  }
#pragma unroll
  for (int off = 16; off; off >>= 1) s += __shfl_xor(s, off, 32);
  if (lane == 0) ssum[wave] = s;
  __syncthreads();
  s = 0.0f;
#pragma unroll
  for (int i = 0; i < 8; ++i) s += ssum[i];
  const float inv = 1.0f / s;
#pragma unroll
  for (int i = 0; i < 8; ++i) row[tid + (i << 8)] = v[i] * inv;
}

// ---------------------------------------------------------------------------
extern "C" void kernel_launch(void* const* d_in, const int* in_sizes, int n_in,
                              void* d_out, int out_size, void* d_ws,
                              size_t ws_size, hipStream_t stream) {
  const char* q     = (const char*)d_in[0];   // [B,QL,QD] f32
  const char* s     = (const char*)d_in[1];   // [B,SL,SD] f32
  const int*  mask  = (const int*)d_in[2];    // [B,1,SL]  i32
  const float* w    = (const float*)d_in[3];  // [QD,SD]   f32
  const char* out_w = (const char*)d_in[4];   // [OD,SD]   f32
  const float* out_b = (const float*)d_in[5]; // [OD]      f32

  const int QL = 2048, SL = 2048, QD = 512, SD = 512, OD = 512;
  const int B = in_sizes[0] / (QL * QD);

  // workspace layout (bf16 buffers)
  char* ws = (char*)d_ws;
  char* wT  = ws;                                       // [SD][QD]
  char* sT  = wT + (size_t)SD * QD * 2;                 // [B][SD][SL]
  char* qw  = sT + (size_t)B * SD * (size_t)SL * 2;     // [B][QL][SD]
  char* mid = qw + (size_t)B * QL * (size_t)SD * 2;     // [B][QL][SD]

  char* attn_w = (char*)d_out;                                     // f32 [B][QL][SL]
  char* attn_o = attn_w + (size_t)B * QL * (size_t)SL * 4;         // f32 [B][QL][OD]

  // 1) w -> wT (bf16), s -> sT (bf16)
  transpose_f32_bf16<<<dim3(SD / 32, QD / 32, 1), 256, 0, stream>>>(
      w, (__bf16*)wT, QD, SD, 0, 0);
  transpose_f32_bf16<<<dim3(SD / 32, SL / 32, B), 256, 0, stream>>>(
      (const float*)s, (__bf16*)sT, SL, SD, (long)SL * SD, (long)SD * SL);

  // 2) qw = q @ w        (A=f32 q, Bt=bf16 wT, out bf16), flat M = B*QL
  gemm_bf16_wmma<true, false, 2><<<dim3(SD / 128, (B * QL) / 128, 1), 256, 0,
                                   stream>>>(
      q, wT, qw, SD, QD, 0, 0, 0, nullptr, 0, nullptr);

  // 3) scores = qw @ s^T  (A=bf16 qw, Bt=f32 s) + mask, out f32 -> attn_w
  gemm_bf16_wmma<false, true, 1><<<dim3(SL / 128, QL / 128, B), 256, 0,
                                   stream>>>(
      qw, s, attn_w, SL, SD, (long)QL * SD, (long)SL * SD, (long)QL * SL,
      mask, SL, nullptr);

  // 4) softmax rows of attn_w in place
  softmax_row2048<<<dim3(B * QL, 1, 1), 256, 0, stream>>>((float*)attn_w);

  // 5) mid = attn_w @ s   (A=f32 attn_w, Bt=bf16 sT), out bf16
  gemm_bf16_wmma<true, false, 2><<<dim3(SD / 128, QL / 128, B), 256, 0,
                                   stream>>>(
      attn_w, sT, mid, SD, SL, (long)QL * SL, (long)SD * SL, (long)QL * SD,
      nullptr, 0, nullptr);

  // 6) attn_out = mid @ out_w^T + out_b (A=bf16 mid, Bt=f32 out_w), out f32
  gemm_bf16_wmma<false, true, 3><<<dim3(OD / 128, (B * QL) / 128, 1), 256, 0,
                                   stream>>>(
      mid, out_w, attn_o, OD, SD, 0, 0, 0, nullptr, 0, out_b);
}
